// GPTLanguageModel_32719060861472
// MI455X (gfx1250) — matrix operand order repaired and checked by reading, not compiled, not run on hardware
//
#include <hip/hip_runtime.h>
#include <hip/hip_bf16.h>
#include <stdint.h>
#include <stddef.h>

// ---------------------------------------------------------------------------
// Model dimensions (fixed by the reference)
// ---------------------------------------------------------------------------
#define Cdim   384
#define Tdim   256
#define Bdim   64
#define Hn     6
#define HDdim  64
#define Mrows  (Bdim * Tdim)   // 16384
#define FFdim  (4 * Cdim)      // 1536
#define Vvocab 65
#define VPAD   128             // LM head N padded to a multiple of 128

typedef __attribute__((ext_vector_type(16))) __bf16 v16bf;
typedef __attribute__((ext_vector_type(8)))  float  v8f;
typedef __attribute__((ext_vector_type(4)))  unsigned int uint4v;

#define WMMA_BF16(a, b, c) \
  __builtin_amdgcn_wmma_f32_16x16x32_bf16(false, (a), false, (b), (short)0, (c), false, false)

// ---------------------------------------------------------------------------
// CDNA5 async global->LDS copy (ASYNCcnt path, CDNA5 ISA).
// Each enabled lane copies 16B from its global address to its LDS address.
// LDS byte offset = low 32 bits of the generic pointer.
// ---------------------------------------------------------------------------
__device__ __forceinline__ unsigned lds_off(const void* p) {
  return (unsigned)(unsigned long long)p;
}
__device__ __forceinline__ void async_copy16(unsigned dst_lds, const void* src) {
  asm volatile("global_load_async_to_lds_b128 %0, %1, off"
               :: "v"(dst_lds), "v"(src) : "memory");
}
#define ASYNC_WAIT_LE4() asm volatile("s_wait_asynccnt 0x4" ::: "memory")
#define ASYNC_WAIT_0()   asm volatile("s_wait_asynccnt 0x0" ::: "memory")

// ---------------------------------------------------------------------------
// WMMA fragment loaders from bf16 LDS tiles.
// A (16x32, MxK): lane = row M (lane&15); half-wave selects K 0-7/16-23 vs
//                 8-15/24-31 (ISA 7.12.2 "16-bit A-Matrix 16x32").
// B (32x16, KxN): lane = col N (lane&15); lanes 0-15 hold K=0-7/16-23, lanes 16-31
//                 hold K=8-15/24-31.  B is staged transposed (N,K) so each lane
//                 reads two contiguous 16B chunks.
// ---------------------------------------------------------------------------
__device__ __forceinline__ v16bf frag_a(const __bf16* base, int ld, int lane) {
  int row = lane & 15;
  int kb  = (lane >> 4) << 3;             // 0 or 8
  union { v16bf v; uint4v u[2]; } f;
  const __bf16* p = base + (size_t)row * ld;
  f.u[0] = *(const uint4v*)(p + kb);
  f.u[1] = *(const uint4v*)(p + 16 + kb);
  return f.v;
}

__device__ __forceinline__ v16bf frag_b(const __bf16* baseT, int ld, int lane) {
  int n  = lane & 15;
  int kb = (lane >> 4) << 3;              // 0 or 8
  union { v16bf v; uint4v u[2]; } f;
  const __bf16* p = baseT + (size_t)n * ld + kb;
  f.u[0] = *(const uint4v*)(p);
  f.u[1] = *(const uint4v*)(p + 16);
  return f.v;
}

// ---------------------------------------------------------------------------
// bf16 WMMA GEMM:  out = [res +] A(MxK) * Bt(N,K)^T [+ bias] [ReLU]
// 128x128 block tile, K-step 32, 8 waves (256 threads); wave owns 32x64 =
// 2x4 wmma tiles.  Double-buffered LDS fed by async global->LDS copies:
// the next K-tile streams in via ASYNCcnt while WMMAs consume the current
// one.  flags: bit0 -> bf16 output, bit1 -> ReLU.  n_clamp masks stores.
// ---------------------------------------------------------------------------
#define GF_OUT_BF16 1
#define GF_RELU     2

__global__ __launch_bounds__(256)
void gemm_bf16_wmma(const __bf16* __restrict__ A, const __bf16* __restrict__ Bt,
                    const float* __restrict__ bias, const float* __restrict__ res,
                    void* __restrict__ out, int K, int ldo, int n_clamp, int flags) {
  __shared__ __align__(16) __bf16 As[2][128 * 32];
  __shared__ __align__(16) __bf16 Bs[2][128 * 32];

  const int m0 = blockIdx.y * 128, n0 = blockIdx.x * 128;
  const int tid  = threadIdx.x;
  const int lane = tid & 31;
  const int wave = tid >> 5;
  const int wm = (wave >> 1) * 32;        // 0,32,64,96
  const int wn = (wave & 1) * 64;         // 0,64

  // Per-thread staging coords: row 0..127, half-row (two 16B chunks).
  const int r  = tid >> 1;
  const int cc = (tid & 1) << 4;          // 0 or 16

  v8f acc[2][4] = {};

  auto issue_tile = [&](int k0, int buf) {
    const __bf16* gA = &A [(size_t)(m0 + r) * K + k0 + cc];
    const __bf16* gB = &Bt[(size_t)(n0 + r) * K + k0 + cc];
    unsigned la = lds_off(&As[buf][r * 32 + cc]);
    unsigned lb = lds_off(&Bs[buf][r * 32 + cc]);
    async_copy16(la,      gA);
    async_copy16(la + 16, gA + 8);
    async_copy16(lb,      gB);
    async_copy16(lb + 16, gB + 8);        // 4 async ops / thread / tile
  };

  const int nk = K >> 5;
  issue_tile(0, 0);
  for (int it = 0; it < nk; ++it) {
    const int curb = it & 1;
    if (it + 1 < nk) {
      issue_tile((it + 1) << 5, curb ^ 1);   // overlap next tile with compute
      ASYNC_WAIT_LE4();                      // in-order: current tile landed
    } else {
      ASYNC_WAIT_0();
    }
    __syncthreads();

#pragma unroll
    for (int i = 0; i < 2; ++i) {
      v16bf a = frag_a(&As[curb][(wm + i * 16) * 32], 32, lane);
#pragma unroll
      for (int j = 0; j < 4; ++j) {
        v16bf b = frag_b(&Bs[curb][(wn + j * 16) * 32], 32, lane);
        acc[i][j] = WMMA_BF16(a, b, acc[i][j]);
      }
    }
    __syncthreads();   // done reading before next iteration overwrites buffer
  }

  // Epilogue.  C/D layout: VGPR v -> M = v + 8*(lane>=16), N = lane&15.
#pragma unroll
  for (int i = 0; i < 2; ++i) {
#pragma unroll
    for (int j = 0; j < 4; ++j) {
      int gn = n0 + wn + j * 16 + (lane & 15);
      if (gn >= n_clamp) continue;
      int mb = m0 + wm + i * 16 + ((lane >> 4) << 3);
      float bs = bias ? bias[gn] : 0.0f;
#pragma unroll
      for (int v = 0; v < 8; ++v) {
        int m = mb + v;
        float val = acc[i][j][v] + bs;
        if (res) val += res[(size_t)m * ldo + gn];
        if (flags & GF_RELU) val = fmaxf(val, 0.0f);
        if (flags & GF_OUT_BF16) ((__bf16*)out)[(size_t)m * ldo + gn] = (__bf16)val;
        else                     ((float*)out)[(size_t)m * ldo + gn] = val;
      }
    }
  }
}

// ---------------------------------------------------------------------------
// Fused causal attention for one (batch, head): the full 256x256 score matrix
// lives in LDS (224KB -- CDNA5's 320KB WGP LDS makes this possible).
//   S = softmax(mask(Q K^T / 8));  O = S V
// Q/K staged with async global->LDS copies; V via VGPRs (needs transpose).
// 8 waves; wave w owns output rows 32w..32w+31.
// ---------------------------------------------------------------------------
#define ATT_LDS_BYTES (Tdim*HDdim*2 /*Qs*/ + Tdim*HDdim*2 /*Ks*/ + \
                       HDdim*Tdim*2 /*Vt*/ + Tdim*Tdim*2 /*S*/)

__global__ __launch_bounds__(256)
void attention_kernel(const __bf16* __restrict__ Q, const __bf16* __restrict__ K,
                      const __bf16* __restrict__ V, __bf16* __restrict__ O) {
  extern __shared__ __align__(16) char smem[];
  __bf16* Qs = (__bf16*)smem;                 // [256][64]
  __bf16* Ks = Qs + Tdim * HDdim;             // [256][64]  (n=s, k=d for B)
  __bf16* Vt = Ks + Tdim * HDdim;             // [64][256]  (n=d, k=s for B)
  __bf16* S  = Vt + HDdim * Tdim;             // [256][256]

  const int bh = blockIdx.x;
  const int b  = bh / Hn, h = bh % Hn;
  const int tid = threadIdx.x, lane = tid & 31, wave = tid >> 5;
  const size_t base = ((size_t)b * Tdim) * Cdim + (size_t)h * HDdim;

  // Stage Q, K (async, no VGPR round-trip) and transposed V (via VGPRs).
  for (int it = 0; it < 8; ++it) {
    int id = it * 256 + tid;                  // 0..2047
    int r  = id >> 3;                         // 0..255
    int d  = (id & 7) << 3;                   // 0,8,..,56
    async_copy16(lds_off(&Qs[r * HDdim + d]), &Q[base + (size_t)r * Cdim + d]);
    async_copy16(lds_off(&Ks[r * HDdim + d]), &K[base + (size_t)r * Cdim + d]);
    uint4v vv = *(const uint4v*)&V[base + (size_t)r * Cdim + d];
    const __bf16* pv = (const __bf16*)&vv;
#pragma unroll
    for (int j = 0; j < 8; ++j) Vt[(d + j) * Tdim + r] = pv[j];
  }
  ASYNC_WAIT_0();
  __syncthreads();

  // Scores: rows 32w..32w+31, 16 column tiles, K=64 (2 wmma per tile).
  const int m0 = wave * 32;
  for (int rt = 0; rt < 2; ++rt) {
    for (int ct = 0; ct < 16; ++ct) {
      v8f acc = {};
#pragma unroll
      for (int ks = 0; ks < 2; ++ks) {
        v16bf a  = frag_a(Qs + (m0 + rt * 16) * HDdim + ks * 32, HDdim, lane);
        v16bf bb = frag_b(Ks + (ct * 16) * HDdim + ks * 32, HDdim, lane);
        acc = WMMA_BF16(a, bb, acc);
      }
      int n  = ct * 16 + (lane & 15);
      int mb = m0 + rt * 16 + ((lane >> 4) << 3);
#pragma unroll
      for (int v = 0; v < 8; ++v)
        S[(size_t)(mb + v) * Tdim + n] = (__bf16)(acc[v] * 0.125f);  // HD^-0.5
    }
  }
  __syncthreads();

  // Causal softmax, one wave per row (wave32 shfl reductions).
  for (int rr = 0; rr < 32; ++rr) {
    int r = m0 + rr;
    float vals[8];
    float mx = -3.0e38f;
#pragma unroll
    for (int ch = 0; ch < 8; ++ch) {
      int c = ch * 32 + lane;
      float s = (c <= r) ? (float)S[(size_t)r * Tdim + c] : -3.0e38f;
      vals[ch] = s;
      mx = fmaxf(mx, s);
    }
#pragma unroll
    for (int o = 16; o; o >>= 1) mx = fmaxf(mx, __shfl_xor(mx, o, 32));
    float sum = 0.0f;
#pragma unroll
    for (int ch = 0; ch < 8; ++ch) {
      int c = ch * 32 + lane;
      float e = (c <= r) ? __expf(vals[ch] - mx) : 0.0f;
      vals[ch] = e;
      sum += e;
    }
#pragma unroll
    for (int o = 16; o; o >>= 1) sum += __shfl_xor(sum, o, 32);
    float inv = 1.0f / sum;
#pragma unroll
    for (int ch = 0; ch < 8; ++ch) {
      int c = ch * 32 + lane;
      S[(size_t)r * Tdim + c] = (__bf16)(vals[ch] * inv);
    }
  }
  __syncthreads();

  // O = S @ V: 2 row-tiles x 4 col-tiles, K=256 (8 wmma each).
  for (int rt = 0; rt < 2; ++rt) {
    for (int ct = 0; ct < 4; ++ct) {
      v8f acc = {};
#pragma unroll
      for (int ks = 0; ks < 8; ++ks) {
        v16bf a  = frag_a(S  + (size_t)(m0 + rt * 16) * Tdim + ks * 32, Tdim, lane);
        v16bf bb = frag_b(Vt + (size_t)(ct * 16) * Tdim + ks * 32, Tdim, lane);
        acc = WMMA_BF16(a, bb, acc);
      }
      int n  = ct * 16 + (lane & 15);
      int mb = m0 + rt * 16 + ((lane >> 4) << 3);
#pragma unroll
      for (int v = 0; v < 8; ++v)
        O[base + (size_t)(mb + v) * Cdim + n] = (__bf16)acc[v];
    }
  }
}

// ---------------------------------------------------------------------------
// Wave-per-row LayerNorm (C=384 = 32 lanes x 12), bf16 output for WMMA.
// ---------------------------------------------------------------------------
__global__ __launch_bounds__(256)
void layernorm_bf16(const float* __restrict__ x, const float* __restrict__ s,
                    const float* __restrict__ bb, __bf16* __restrict__ o) {
  int row  = blockIdx.x * 8 + (threadIdx.x >> 5);
  int lane = threadIdx.x & 31;
  const float* xr = x + (size_t)row * Cdim;
  float vals[12];
  float sum = 0.0f, sq = 0.0f;
#pragma unroll
  for (int i = 0; i < 12; ++i) {
    float v = xr[lane + 32 * i];
    vals[i] = v; sum += v; sq += v * v;
  }
#pragma unroll
  for (int o2 = 16; o2; o2 >>= 1) {
    sum += __shfl_xor(sum, o2, 32);
    sq  += __shfl_xor(sq,  o2, 32);
  }
  float mean = sum * (1.0f / Cdim);
  float var  = sq * (1.0f / Cdim) - mean * mean;
  float rstd = rsqrtf(var + 1e-5f);
  __bf16* orow = o + (size_t)row * Cdim;
#pragma unroll
  for (int i = 0; i < 12; ++i) {
    int c = lane + 32 * i;
    orow[c] = (__bf16)((vals[i] - mean) * rstd * s[c] + bb[c]);
  }
}

// ---------------------------------------------------------------------------
// Embedding gather: x[b,t,:] = tok_emb[idx[b,t],:] + pos_emb[t,:]
// ---------------------------------------------------------------------------
__global__ void embed_kernel(const float* __restrict__ tok, const float* __restrict__ pos,
                             const int* __restrict__ idx, float* __restrict__ x) {
  int id  = blockIdx.x * blockDim.x + threadIdx.x;     // < Mrows*Cdim
  int row = id / Cdim, c = id - row * Cdim;
  int t   = row & (Tdim - 1);
  x[id] = tok[(size_t)idx[row] * Cdim + c] + pos[(size_t)t * Cdim + c];
}

// ---------------------------------------------------------------------------
// Weight packing: row-major (K,N) f32 -> transposed (Npad,K) bf16 (zero pad)
// ---------------------------------------------------------------------------
__global__ void pack_nk(const float* __restrict__ W, __bf16* __restrict__ Bt,
                        int K, int N, int Npad) {
  int id = blockIdx.x * 256 + threadIdx.x;
  if (id >= Npad * K) return;
  int n = id / K, k = id - n * K;
  Bt[id] = (n < N) ? (__bf16)W[(size_t)k * N + n] : (__bf16)0.0f;
}

// (H,C,HD) f32 -> (N=H*HD, K=C) bf16 so q[bt, h*64+d] = sum_c h[bt,c] wq[h,c,d]
__global__ void pack_qkv(const float* __restrict__ W, __bf16* __restrict__ Bt) {
  int id = blockIdx.x * 256 + threadIdx.x;             // < 384*384
  int n = id / Cdim, k = id - n * Cdim;                // n = h*64 + d
  int h = n >> 6, d = n & 63;
  Bt[id] = (__bf16)W[((size_t)h * Cdim + k) * HDdim + d];
}

// ---------------------------------------------------------------------------
// Gumbel gating (deterministic hash RNG), wave per row.
//   x <- g*mu + (1-g)*x ;  prevGate <- g (ANDed with previous layer's gate)
// ---------------------------------------------------------------------------
__device__ __forceinline__ float hash_uniform(unsigned a) {
  a ^= a >> 16; a *= 0x7feb352du;
  a ^= a >> 15; a *= 0x846ca68bu;
  a ^= a >> 16;
  return ((float)(a >> 8) + 0.5f) * (1.0f / 16777216.0f);
}

__global__ __launch_bounds__(256)
void gated_combine(float* __restrict__ x, const float* __restrict__ mu,
                   float* __restrict__ prevGate,
                   const float* __restrict__ gw, const float* __restrict__ gb,
                   const float* __restrict__ lvq_p, int layer, int first) {
  int row  = blockIdx.x * 8 + (threadIdx.x >> 5);
  int lane = threadIdx.x & 31;
  float*       xr = x  + (size_t)row * Cdim;
  const float* mr = mu + (size_t)row * Cdim;
  float l2 = 0.0f;
#pragma unroll
  for (int i = 0; i < 12; ++i) {
    float d = mr[lane + 32 * i] - xr[lane + 32 * i];
    l2 += d * d;
  }
#pragma unroll
  for (int o = 16; o; o >>= 1) l2 += __shfl_xor(l2, o, 32);
  float lvq = lvq_p[0];
  float surprise = l2 * 0.5f * __expf(-lvq) - 0.5f * lvq;
  float z  = __expf(gw[0]) * surprise - gb[0];
  float pi = 1.0f / (1.0f + __expf(-z));
  float u  = hash_uniform(((unsigned)layer * 1315423911u) ^
                          (((unsigned)row * 2654435761u) + 0x9e3779b9u));
  float g  = -__logf(-__logf(u));                     // Gumbel noise
  float gate = (__logf(pi) + g > __logf(1.0f - pi)) ? 1.0f : 0.0f;
  if (!first) gate *= prevGate[row];
#pragma unroll
  for (int i = 0; i < 12; ++i) {
    int c = lane + 32 * i;
    xr[c] = gate * mr[c] + (1.0f - gate) * xr[c];
  }
  if (lane == 0) prevGate[row] = gate;
}

// ---------------------------------------------------------------------------
// Host-side orchestration
// ---------------------------------------------------------------------------
struct BlockW {
  const float *wq, *wk, *wv, *proj_w, *proj_b, *w1, *b1, *w2, *b2,
              *ln1s, *ln1b, *ln2s, *ln2b;
};
struct PackedW { __bf16 *wq, *wk, *wv, *proj, *w1, *w2; };

extern "C" void kernel_launch(void* const* d_in, const int* in_sizes, int n_in,
                              void* d_out, int out_size, void* d_ws, size_t ws_size,
                              hipStream_t stream) {
  (void)in_sizes; (void)n_in; (void)out_size; (void)ws_size;

  // ---- input mapping (setup_inputs() dict order, depth-first) ----
  int gi = 0;
  auto nextf = [&]() { return (const float*)d_in[gi++]; };
  auto readBlock = [&]() {
    BlockW w;
    w.wq = nextf(); w.wk = nextf(); w.wv = nextf();
    w.proj_w = nextf(); w.proj_b = nextf();
    w.w1 = nextf(); w.b1 = nextf(); w.w2 = nextf(); w.b2 = nextf();
    w.ln1s = nextf(); w.ln1b = nextf(); w.ln2s = nextf(); w.ln2b = nextf();
    return w;
  };
  const float* tok_emb = nextf();
  const float* pos_emb = nextf();
  BlockW blocks[6];
  const float *gw[4], *gb[4], *lvq[4];
  blocks[0] = readBlock();                       // entry
  for (int i = 0; i < 4; ++i) {                  // gated
    blocks[1 + i] = readBlock();
    gw[i] = nextf(); gb[i] = nextf(); lvq[i] = nextf();
  }
  blocks[5] = readBlock();                       // exit
  const float* ln_f_s = nextf();
  const float* ln_f_b = nextf();
  const float* lm_w   = nextf();
  const float* lm_b   = nextf();
  const int*   idx    = (const int*)d_in[gi++];

  // ---- workspace carve ----
  char* wp = (char*)d_ws;
  auto carve = [&](size_t bytes) {
    void* r = (void*)wp;
    wp += (bytes + 255) & ~(size_t)255;
    return r;
  };
  float*  B1  = (float*)carve((size_t)Mrows * Cdim * 4);
  float*  B2  = (float*)carve((size_t)Mrows * Cdim * 4);
  __bf16* Hb  = (__bf16*)carve((size_t)Mrows * Cdim * 2);
  __bf16* Qb  = (__bf16*)carve((size_t)Mrows * Cdim * 2);
  __bf16* Kb  = (__bf16*)carve((size_t)Mrows * Cdim * 2);
  __bf16* Vb  = (__bf16*)carve((size_t)Mrows * Cdim * 2);
  __bf16* Ab  = (__bf16*)carve((size_t)Mrows * Cdim * 2);
  __bf16* FFb = (__bf16*)carve((size_t)Mrows * FFdim * 2);
  float*  prevGate = (float*)carve((size_t)Mrows * 4);
  PackedW pk[6];
  for (int i = 0; i < 6; ++i) {
    pk[i].wq   = (__bf16*)carve((size_t)Cdim * Cdim * 2);
    pk[i].wk   = (__bf16*)carve((size_t)Cdim * Cdim * 2);
    pk[i].wv   = (__bf16*)carve((size_t)Cdim * Cdim * 2);
    pk[i].proj = (__bf16*)carve((size_t)Cdim * Cdim * 2);
    pk[i].w1   = (__bf16*)carve((size_t)FFdim * Cdim * 2);
    pk[i].w2   = (__bf16*)carve((size_t)Cdim * FFdim * 2);
  }
  __bf16* lmT = (__bf16*)carve((size_t)VPAD * Cdim * 2);

  // ---- pack all weights to bf16 (N,K) ----
  for (int i = 0; i < 6; ++i) {
    pack_qkv<<<(Cdim * Cdim) / 256, 256, 0, stream>>>(blocks[i].wq, pk[i].wq);
    pack_qkv<<<(Cdim * Cdim) / 256, 256, 0, stream>>>(blocks[i].wk, pk[i].wk);
    pack_qkv<<<(Cdim * Cdim) / 256, 256, 0, stream>>>(blocks[i].wv, pk[i].wv);
    pack_nk<<<(Cdim * Cdim) / 256, 256, 0, stream>>>(blocks[i].proj_w, pk[i].proj, Cdim, Cdim, Cdim);
    pack_nk<<<(FFdim * Cdim) / 256, 256, 0, stream>>>(blocks[i].w1, pk[i].w1, Cdim, FFdim, FFdim);
    pack_nk<<<(Cdim * FFdim) / 256, 256, 0, stream>>>(blocks[i].w2, pk[i].w2, FFdim, Cdim, Cdim);
  }
  pack_nk<<<(VPAD * Cdim) / 256, 256, 0, stream>>>(lm_w, lmT, Cdim, Vvocab, VPAD);

  // ---- embedding ----
  embed_kernel<<<(Mrows * Cdim) / 256, 256, 0, stream>>>(tok_emb, pos_emb, idx, B1);

  const dim3 gC(Cdim / 128, Mrows / 128);   // N=384 GEMMs  (3 x 128)
  const dim3 gF(FFdim / 128, Mrows / 128);  // N=1536 GEMM  (12 x 128)
  const dim3 gV(VPAD / 128, Mrows / 128);   // LM head      (1 x 128)

  auto runBlock = [&](const BlockW& w, const PackedW& q, const float* xin, float* xout) {
    layernorm_bf16<<<Mrows / 8, 256, 0, stream>>>(xin, w.ln1s, w.ln1b, Hb);
    gemm_bf16_wmma<<<gC, 256, 0, stream>>>(Hb, q.wq, nullptr, nullptr, Qb, Cdim, Cdim, Cdim, GF_OUT_BF16);
    gemm_bf16_wmma<<<gC, 256, 0, stream>>>(Hb, q.wk, nullptr, nullptr, Kb, Cdim, Cdim, Cdim, GF_OUT_BF16);
    gemm_bf16_wmma<<<gC, 256, 0, stream>>>(Hb, q.wv, nullptr, nullptr, Vb, Cdim, Cdim, Cdim, GF_OUT_BF16);
    attention_kernel<<<Bdim * Hn, 256, ATT_LDS_BYTES, stream>>>(Qb, Kb, Vb, Ab);
    gemm_bf16_wmma<<<gC, 256, 0, stream>>>(Ab, q.proj, w.proj_b, xin, xout, Cdim, Cdim, Cdim, 0);
    layernorm_bf16<<<Mrows / 8, 256, 0, stream>>>(xout, w.ln2s, w.ln2b, Hb);
    gemm_bf16_wmma<<<gF, 256, 0, stream>>>(Hb, q.w1, w.b1, nullptr, FFb, Cdim, FFdim, FFdim,
                                           GF_OUT_BF16 | GF_RELU);
    gemm_bf16_wmma<<<gC, 256, 0, stream>>>(FFb, q.w2, w.b2, xout, xout, FFdim, Cdim, Cdim, 0);
  };

  // entry block (in-place on B1)
  runBlock(blocks[0], pk[0], B1, B1);
  // gated blocks: mu into B2, then stochastic combine back into B1
  for (int i = 0; i < 4; ++i) {
    runBlock(blocks[1 + i], pk[1 + i], B1, B2);
    gated_combine<<<Mrows / 8, 256, 0, stream>>>(B1, B2, prevGate, gw[i], gb[i], lvq[i], i, i == 0);
  }
  // exit block (in-place on B1)
  runBlock(blocks[5], pk[5], B1, B1);

  // final LN + LM head (N padded to 128, stores clamped to 65, ldo=65)
  layernorm_bf16<<<Mrows / 8, 256, 0, stream>>>(B1, ln_f_s, ln_f_b, Hb);
  gemm_bf16_wmma<<<gV, 256, 0, stream>>>(Hb, lmT, lm_b, nullptr, d_out, Cdim, Vvocab, Vvocab, 0);
}
